// DKVMN_41944650612757
// MI455X (gfx1250) — matrix operand order repaired and checked by reading, not compiled, not run on hardware
//
#include <hip/hip_runtime.h>
#include <hip/hip_bf16.h>
#include <cstdint>

#define NUM_C 128
#define DIM_S 128
#define SIZE_M 50
#define BB 64
#define TT 200
#define ROWS (BB * TT)   // 12800, divisible by 16 -> 800 full WMMA row tiles
#define MPAD 64          // padded N for the w/softmax GEMM (50 -> 64)

typedef __attribute__((ext_vector_type(16))) _Float16 v16h;
typedef __attribute__((ext_vector_type(8)))  _Float16 v8h;
typedef __attribute__((ext_vector_type(8)))  float    v8f;
typedef __attribute__((ext_vector_type(2)))  int      v2i;

union V16 { v16h v; v8h h[2]; };

// 16-bit A matrix 16x32 fragment: lane m=lane&15, half_sel=lane>>4.
// elements 0..7  = A[m][base + hs*8 + 0..7]
// elements 8..15 = A[m][base + 16 + hs*8 + 0..7]
__device__ inline v16h load_a16(const _Float16* rowp, int off) {
  V16 u;
  u.h[0] = *(const v8h*)(rowp + off);
  u.h[1] = *(const v8h*)(rowp + off + 16);
  return u.v;
}
// 16-bit B matrix 32x16 fragment from transposed (N-major) weight row:
// elements 0..7 = B[base + hs*8 + 0..7][n], elements 8..15 = B[base + 16 + hs*8 + 0..7][n]
__device__ inline v16h load_b16(const _Float16* rowp, int off) {
  V16 u;
  u.h[0] = *(const v8h*)(rowp + off);
  u.h[1] = *(const v8h*)(rowp + off + 16);
  return u.v;
}
__device__ inline v8f wmma_f16(v16h a, v16h b, v8f c) {
  return __builtin_amdgcn_wmma_f32_16x16x32_f16(false, a, false, b, (short)0, c,
                                                false, false);
}
__device__ inline float sigmoidf_(float x) { return 1.0f / (1.0f + __expf(-x)); }

// ---- gfx1250 async global->LDS DMA (ASYNCcnt-tracked), guarded fallback ----
#if __has_builtin(__builtin_amdgcn_global_load_async_to_lds_b64)
#define HAVE_ASYNC_LDS 1
typedef __attribute__((address_space(1))) v2i g_v2i;
typedef __attribute__((address_space(3))) v2i l_v2i;
__device__ inline void async_copy_b64(const void* g, void* lds_generic) {
  __builtin_amdgcn_global_load_async_to_lds_b64(
      (g_v2i*)(uintptr_t)g,
      (l_v2i*)(uint32_t)(uintptr_t)lds_generic,
      0, 0);
}
__device__ inline void wait_async0() {
#if __has_builtin(__builtin_amdgcn_s_wait_asynccnt)
  __builtin_amdgcn_s_wait_asynccnt(0);
#else
  asm volatile("s_wait_asynccnt 0x0" ::: "memory");
#endif
}
#else
#define HAVE_ASYNC_LDS 0
#endif

// ---------------- prep: f16 transposed weights ----------------
__global__ void prep_weights(const float* __restrict__ Mk, const float* __restrict__ We,
                             const float* __restrict__ Wa, const float* __restrict__ Wf,
                             _Float16* __restrict__ MkH, _Float16* __restrict__ WeT,
                             _Float16* __restrict__ WaT, _Float16* __restrict__ WfT) {
  int i = blockIdx.x * blockDim.x + threadIdx.x;
  if (i < MPAD * DIM_S) {                       // MkH[m][d], zero-padded rows 50..63
    int m = i >> 7, d = i & 127;
    MkH[i] = (m < SIZE_M) ? (_Float16)Mk[m * DIM_S + d] : (_Float16)0.0f;
  }
  if (i < DIM_S * DIM_S) {                      // WeT[n][d] = We[d][n]
    int n = i >> 7, d = i & 127;
    WeT[i] = (_Float16)We[d * DIM_S + n];
    WaT[i] = (_Float16)Wa[d * DIM_S + n];
  }
  if (i < 2 * DIM_S * DIM_S) {                  // WfT[n][c] = Wf[c][n], c in [0,256)
    int n = i >> 8, c = i & 255;
    WfT[i] = (_Float16)Wf[c * DIM_S + n];
  }
}

// ---------------- gather embeddings to f16 ----------------
__global__ void gather_kv(const int* __restrict__ q, const int* __restrict__ r,
                          const float* __restrict__ k_emb, const float* __restrict__ v_emb,
                          _Float16* __restrict__ khalf, _Float16* __restrict__ vhalf) {
  int i = blockIdx.x * blockDim.x + threadIdx.x;   // ROWS*128 threads
  int row = i >> 7, d = i & 127;
  int qv = q[row], rv = r[row];
  khalf[i] = (_Float16)k_emb[qv * DIM_S + d];
  vhalf[i] = (_Float16)v_emb[(qv + NUM_C * rv) * DIM_S + d];
}

// ---------------- stage 1: w = softmax(k Mk^T), e = sigmoid(v We), a = tanh(v Wa)
__global__ __launch_bounds__(256) void gemm_stage1(
    const _Float16* __restrict__ khalf, const _Float16* __restrict__ vhalf,
    const _Float16* __restrict__ MkH, const _Float16* __restrict__ WeT,
    const _Float16* __restrict__ WaT, const float* __restrict__ be,
    const float* __restrict__ ba, float* __restrict__ wbuf,
    _Float16* __restrict__ eH, _Float16* __restrict__ aH) {
  const int lane = threadIdx.x & 31;
  const int wave = threadIdx.x >> 5;
  const int tile = blockIdx.x * 8 + wave;       // 100 blocks * 8 waves = 800 tiles
  const int rowBase = tile * 16;
  const int mrow = lane & 15;
  const int hs = lane >> 4;

  // ---- logits (16x64) over K=128, then row softmax ----
  v16h ak[4];
  {
    const _Float16* arow = khalf + (size_t)(rowBase + mrow) * DIM_S;
#pragma unroll
    for (int kt = 0; kt < 4; ++kt) ak[kt] = load_a16(arow, kt * 32 + hs * 8);
  }
  v8f lg[4];
#pragma unroll
  for (int n = 0; n < 4; ++n) {
    v8f c = {};
    const _Float16* brow = MkH + (size_t)(mrow + 16 * n) * DIM_S;
#pragma unroll
    for (int kt = 0; kt < 4; ++kt) c = wmma_f16(ak[kt], load_b16(brow, kt * 32 + hs * 8), c);
    lg[n] = c;
  }
  // mask padded columns (>= 50) before softmax
#pragma unroll
  for (int n = 0; n < 4; ++n) {
    int col = mrow + 16 * n;
    if (col >= SIZE_M) {
#pragma unroll
      for (int rr = 0; rr < 8; ++rr) lg[n][rr] = -1e30f;
    }
  }
  // per-row (M) reductions: in-lane over n, then xor-shuffle over the 16-lane group
#pragma unroll
  for (int rr = 0; rr < 8; ++rr) {
    float mx = lg[0][rr];
#pragma unroll
    for (int n = 1; n < 4; ++n) mx = fmaxf(mx, lg[n][rr]);
#pragma unroll
    for (int m = 1; m <= 8; m <<= 1) mx = fmaxf(mx, __shfl_xor(mx, m, 32));
    float s = 0.f;
#pragma unroll
    for (int n = 0; n < 4; ++n) { float ev = __expf(lg[n][rr] - mx); lg[n][rr] = ev; s += ev; }
#pragma unroll
    for (int m = 1; m <= 8; m <<= 1) s += __shfl_xor(s, m, 32);
    float inv = 1.0f / s;
#pragma unroll
    for (int n = 0; n < 4; ++n)
      wbuf[(size_t)(rowBase + rr + 8 * hs) * MPAD + (mrow + 16 * n)] = lg[n][rr] * inv;
  }

  // ---- e / a: 16x128 over K=128 ----
  v16h av[4];
  {
    const _Float16* arow = vhalf + (size_t)(rowBase + mrow) * DIM_S;
#pragma unroll
    for (int kt = 0; kt < 4; ++kt) av[kt] = load_a16(arow, kt * 32 + hs * 8);
  }
#pragma unroll
  for (int n = 0; n < 8; ++n) {
    const int col = mrow + 16 * n;
    const _Float16* browE = WeT + (size_t)col * DIM_S;
    const _Float16* browA = WaT + (size_t)col * DIM_S;
    v8f ce = {}, ca = {};
#pragma unroll
    for (int kt = 0; kt < 4; ++kt) {
      ce = wmma_f16(av[kt], load_b16(browE, kt * 32 + hs * 8), ce);
      ca = wmma_f16(av[kt], load_b16(browA, kt * 32 + hs * 8), ca);
    }
    const float biasE = be[col], biasA = ba[col];
#pragma unroll
    for (int rr = 0; rr < 8; ++rr) {
      const int row = rowBase + rr + 8 * hs;
      eH[(size_t)row * DIM_S + col] = (_Float16)sigmoidf_(ce[rr] + biasE);
      aH[(size_t)row * DIM_S + col] = (_Float16)tanhf(ca[rr] + biasA);
    }
  }
}

// ---------------- batch-parallel recurrent scan; Mv lives in LDS ----------------
// Per step stages 768 B (w:64 f32, e/a:128 f16). With the async path, wave 0
// DMA-fills the t+1 buffers (3 x b64 per lane, ASYNCcnt) while all 8 waves
// compute step t; s_wait_asynccnt 0 + barrier publishes a buffer.
__global__ __launch_bounds__(256) void scan_steps(
    const float* __restrict__ wbuf, const _Float16* __restrict__ eH,
    const _Float16* __restrict__ aH, const float* __restrict__ Mv0,
    _Float16* __restrict__ readsH) {
  __shared__ float    Mv[SIZE_M * DIM_S];   // 25.6 KB of the 320 KB WGP LDS
  __shared__ float    wsb[2][MPAD];
  __shared__ _Float16 esb[2][DIM_S];
  __shared__ _Float16 asb[2][DIM_S];
  const int b = blockIdx.x;
  const int tid = threadIdx.x;
  for (int i = tid; i < SIZE_M * DIM_S; i += 256) Mv[i] = Mv0[i];

#if HAVE_ASYNC_LDS
  if (tid < 32) {                       // prime buffer 0 for t = 0
    const size_t row0 = (size_t)b * TT;
    async_copy_b64(wbuf + row0 * MPAD + tid * 2, &wsb[0][tid * 2]);
    async_copy_b64(eH + row0 * DIM_S + tid * 4, &esb[0][tid * 4]);
    async_copy_b64(aH + row0 * DIM_S + tid * 4, &asb[0][tid * 4]);
  }
#endif
  __syncthreads();

  for (int t = 0; t < TT; ++t) {
    const int row = b * TT + t;
    const int cur = t & 1, nxt = cur ^ 1;

#if HAVE_ASYNC_LDS
    if (tid < 32) wait_async0();        // buffer[cur] DMA complete (wave 0 issued)
    __syncthreads();                    // publish buffer[cur] to all waves
    if (t + 1 < TT && tid < 32) {       // overlap: DMA-fill buffer[nxt] for t+1
      const size_t rown = (size_t)row + 1;
      async_copy_b64(wbuf + rown * MPAD + tid * 2, &wsb[nxt][tid * 2]);
      async_copy_b64(eH + rown * DIM_S + tid * 4, &esb[nxt][tid * 4]);
      async_copy_b64(aH + rown * DIM_S + tid * 4, &asb[nxt][tid * 4]);
    }
#else
    (void)nxt;
    if (tid < MPAD) wsb[cur][tid] = wbuf[(size_t)row * MPAD + tid];
    if (tid < DIM_S) esb[cur][tid] = eH[(size_t)row * DIM_S + tid];
    else             asb[cur][tid - DIM_S] = aH[(size_t)row * DIM_S + (tid - DIM_S)];
    if (t + 1 < TT && tid == 0) {       // latency hint (global_prefetch_b8)
      __builtin_prefetch(wbuf + (size_t)(row + 1) * MPAD, 0, 0);
      __builtin_prefetch(eH + (size_t)(row + 1) * DIM_S, 0, 0);
      __builtin_prefetch(aH + (size_t)(row + 1) * DIM_S, 0, 0);
    }
    __syncthreads();
#endif

    if (tid < DIM_S) {                  // read BEFORE update (matches reference)
      float s = 0.f;
#pragma unroll
      for (int m = 0; m < SIZE_M; ++m) s = fmaf(wsb[cur][m], Mv[m * DIM_S + tid], s);
      readsH[(size_t)row * DIM_S + tid] = (_Float16)s;
    }
    __syncthreads();
    for (int i = tid; i < SIZE_M * DIM_S; i += 256) {
      const int m = i >> 7, d = i & 127;
      const float wm = wsb[cur][m];
      Mv[i] = Mv[i] * (1.0f - wm * (float)esb[cur][d]) + wm * (float)asb[cur][d];
    }
    __syncthreads();
  }
}

// ---------------- output: f = tanh([read,k] Wf + bf); p = sigmoid(f Wp + bp)
__global__ __launch_bounds__(256) void gemm_out(
    const _Float16* __restrict__ readsH, const _Float16* __restrict__ khalf,
    const _Float16* __restrict__ WfT, const float* __restrict__ bf,
    const float* __restrict__ Wp, const float* __restrict__ bp,
    float* __restrict__ out) {
  const int lane = threadIdx.x & 31;
  const int wave = threadIdx.x >> 5;
  const int tile = blockIdx.x * 8 + wave;
  const int rowBase = tile * 16;
  const int mrow = lane & 15;
  const int hs = lane >> 4;

  v16h afr[8];   // K=256: kt 0..3 from read, 4..7 from k
  {
    const _Float16* r0 = readsH + (size_t)(rowBase + mrow) * DIM_S;
    const _Float16* r1 = khalf + (size_t)(rowBase + mrow) * DIM_S;
#pragma unroll
    for (int kt = 0; kt < 4; ++kt) {
      afr[kt]     = load_a16(r0, kt * 32 + hs * 8);
      afr[kt + 4] = load_a16(r1, kt * 32 + hs * 8);
    }
  }
  float pacc[8];
#pragma unroll
  for (int rr = 0; rr < 8; ++rr) pacc[rr] = 0.f;
#pragma unroll
  for (int n = 0; n < 8; ++n) {
    const int col = mrow + 16 * n;
    const _Float16* brow = WfT + (size_t)col * (2 * DIM_S);
    v8f c = {};
#pragma unroll
    for (int kt = 0; kt < 8; ++kt) c = wmma_f16(afr[kt], load_b16(brow, kt * 32 + hs * 8), c);
    const float bias = bf[col], wp = Wp[col];
#pragma unroll
    for (int rr = 0; rr < 8; ++rr)
      pacc[rr] = fmaf(tanhf(c[rr] + bias), wp, pacc[rr]);   // fold f @ Wp in-register
  }
  const float bpv = bp[0];
#pragma unroll
  for (int rr = 0; rr < 8; ++rr) {
    float v = pacc[rr];
#pragma unroll
    for (int m = 1; m <= 8; m <<= 1) v += __shfl_xor(v, m, 32);   // sum over 16 N-lanes
    if (mrow == 0)
      out[rowBase + rr + 8 * hs] = sigmoidf_(v + bpv);
  }
}

extern "C" void kernel_launch(void* const* d_in, const int* in_sizes, int n_in,
                              void* d_out, int out_size, void* d_ws, size_t ws_size,
                              hipStream_t stream) {
  (void)in_sizes; (void)n_in; (void)out_size; (void)ws_size;
  const int*   q     = (const int*)d_in[0];
  const int*   r     = (const int*)d_in[1];
  const float* k_emb = (const float*)d_in[2];
  const float* v_emb = (const float*)d_in[3];
  const float* Mk    = (const float*)d_in[4];
  const float* Mv0   = (const float*)d_in[5];
  const float* We    = (const float*)d_in[6];
  const float* be    = (const float*)d_in[7];
  const float* Wa    = (const float*)d_in[8];
  const float* ba    = (const float*)d_in[9];
  const float* Wf    = (const float*)d_in[10];
  const float* bf    = (const float*)d_in[11];
  const float* Wp    = (const float*)d_in[12];
  const float* bp    = (const float*)d_in[13];

  char* ws = (char*)d_ws;
  size_t off = 0;
  auto alloc = [&](size_t bytes) -> void* {
    void* p = ws + off;
    off = (off + bytes + 255) & ~(size_t)255;
    return p;
  };
  _Float16* khalf  = (_Float16*)alloc((size_t)ROWS * DIM_S * 2);
  _Float16* vhalf  = (_Float16*)alloc((size_t)ROWS * DIM_S * 2);
  _Float16* readsH = (_Float16*)alloc((size_t)ROWS * DIM_S * 2);
  _Float16* eH     = (_Float16*)alloc((size_t)ROWS * DIM_S * 2);
  _Float16* aH     = (_Float16*)alloc((size_t)ROWS * DIM_S * 2);
  float*    wbuf   = (float*)   alloc((size_t)ROWS * MPAD * 4);
  _Float16* MkH    = (_Float16*)alloc((size_t)MPAD * DIM_S * 2);
  _Float16* WeT    = (_Float16*)alloc((size_t)DIM_S * DIM_S * 2);
  _Float16* WaT    = (_Float16*)alloc((size_t)DIM_S * DIM_S * 2);
  _Float16* WfT    = (_Float16*)alloc((size_t)2 * DIM_S * DIM_S * 2);
  // total workspace ~19.8 MB

  prep_weights<<<128, 256, 0, stream>>>(Mk, We, Wa, Wf, MkH, WeT, WaT, WfT);
  gather_kv<<<(ROWS * DIM_S) / 256, 256, 0, stream>>>(q, r, k_emb, v_emb, khalf, vhalf);
  gemm_stage1<<<ROWS / 16 / 8, 256, 0, stream>>>(khalf, vhalf, MkH, WeT, WaT, be, ba,
                                                 wbuf, eH, aH);
  scan_steps<<<BB, 256, 0, stream>>>(wbuf, eH, aH, Mv0, readsH);
  gemm_out<<<ROWS / 16 / 8, 256, 0, stream>>>(readsH, khalf, WfT, bf, Wp, bp,
                                              (float*)d_out);
}
